// CausalSelfAttention_88562225643803
// MI455X (gfx1250) — matrix operand order repaired and checked by reading, not compiled, not run on hardware
//
#include <hip/hip_runtime.h>

typedef __attribute__((ext_vector_type(16))) __bf16 bf16x16;
typedef __attribute__((ext_vector_type(4)))  __bf16 bf16x4;
typedef __attribute__((ext_vector_type(8)))  float  f32x8;
typedef __attribute__((ext_vector_type(4)))  unsigned int u32x4;
typedef __attribute__((ext_vector_type(8)))  int i32x8;
typedef __attribute__((ext_vector_type(4)))  int i32x4;

namespace {
constexpr int BB = 4;
constexpr int TT = 2048;
constexpr int CC = 1024;
constexpr int HH = 16;
constexpr int DD = 64;
constexpr int MM = BB * TT;   // 8192 rows of x
constexpr int C3 = 3 * CC;    // 3072
}

union FragU {
  bf16x16 v;
  uint4   u[2];
};

__device__ __forceinline__ f32x8 wmma_bf16(bf16x16 a, bf16x16 b, f32x8 c) {
  // D = A(16x32 bf16) * B(32x16 bf16) + C(16x16 f32)
  return __builtin_amdgcn_wmma_f32_16x16x32_bf16(false, a, false, b, (short)0, c,
                                                 false, false);
}

// Issue a TDM load of a [rows x 32] bf16 tile (row-major, row stride
// `stride_elems` bf16 elements) from global into LDS at byte offset lds_addr.
// D# layout per CDNA5 ISA ch.8: group0 = {count, lds_addr, global_addr, type},
// group1 = {flags/data_size, tensor dims, tile dims, strides}.
__device__ __forceinline__ void tdm_load_tile32(unsigned lds_addr,
                                                const void* gptr,
                                                unsigned rows,
                                                unsigned stride_elems) {
  unsigned long long ga = (unsigned long long)(uintptr_t)gptr;
  u32x4 g0 = {1u,                                   // count=1, user descriptor
              lds_addr,                             // LDS byte address
              (unsigned)ga,                         // global_addr[31:0]
              (unsigned)((ga >> 32) & 0x01ffffffu)  // global_addr[56:32]
                  | (2u << 30)};                    // type=2 ("image")
  i32x8 g1 = {(int)(1u << 16),                      // data_size=1 -> 2 bytes
              (int)((stride_elems & 0xffffu) << 16),        // tensor_dim0 lo16
              (int)((stride_elems >> 16) | (rows << 16)),   // dim0 hi | dim1 lo
              (int)(32u << 16),                     // dim1 hi | tile_dim0=32
              (int)rows,                            // tile_dim1 | tile_dim2=0
              (int)stride_elems,                    // tensor_dim0_stride lo32
              0, 0};                                // stride hi / dim1_stride
  i32x4 z4 = {0, 0, 0, 0};
  i32x8 z8 = {0, 0, 0, 0, 0, 0, 0, 0};
  __builtin_amdgcn_tensor_load_to_lds(g0, g1, z4, z4, z8, 0);
}

// ---------------------------------------------------------------- stage 1
__global__ __launch_bounds__(256) void k_cvt_bf16(const float* __restrict__ in,
                                                  __bf16* __restrict__ out, int n) {
  int i = (blockIdx.x * 256 + threadIdx.x) * 4;
  if (i + 3 < n) {
    float4 f = *(const float4*)(in + i);
    bf16x4 o;
    o[0] = (__bf16)f.x; o[1] = (__bf16)f.y; o[2] = (__bf16)f.z; o[3] = (__bf16)f.w;
    *(bf16x4*)(out + i) = o;
  } else {
    for (; i < n; ++i) out[i] = (__bf16)in[i];
  }
}

// ---------------------------------------------------------------- stage 2
// qkv = x @ w_attn^T + b_attn, scattered to Q[b,h,t,d], K[b,h,t,d], Vt[b,h,d,t].
// Block = 128 rows x 64 cols; 8 waves each own a 16x64 register tile. The
// shared 64x32 B tile is TDM double-buffered in LDS (one DMA per block per
// k-step instead of 8 redundant global reloads).
__global__ __launch_bounds__(256) void k_qkv_gemm(
    const __bf16* __restrict__ xb, const __bf16* __restrict__ wb,
    const float* __restrict__ bias,
    __bf16* __restrict__ qO, __bf16* __restrict__ kO, __bf16* __restrict__ vtO) {
  __shared__ __align__(16) __bf16 smemB[2][64 * 32];
  const int lane  = threadIdx.x & 31;
  const int wslot = threadIdx.x >> 5;
  const int nGroups = C3 / 64;              // 48
  const int mBlk = blockIdx.x / nGroups;
  const int n0   = (blockIdx.x % nGroups) * 64;
  const int mIdx = mBlk * 8 + wslot;
  const int hf = lane >> 4;
  const int ln = lane & 15;
  const int rowA = mIdx * 16 + ln;

  const __bf16* wTile = wb + (size_t)n0 * CC;
  if (wslot == 0)
    tdm_load_tile32((unsigned)(uintptr_t)&smemB[0][0], wTile, 64u, (unsigned)CC);

  f32x8 acc[4] = {};
  int p = 0;
  constexpr int nIter = CC / 32;            // 32
  for (int it = 0; it < nIter; ++it) {
    const int kk0 = it * 32;
    if (wslot == 0) {
      if (it + 1 < nIter) {
        tdm_load_tile32((unsigned)(uintptr_t)&smemB[p ^ 1][0], wTile + kk0 + 32,
                        64u, (unsigned)CC);
        __builtin_amdgcn_s_wait_tensorcnt((short)1);
      } else {
        __builtin_amdgcn_s_wait_tensorcnt((short)0);
      }
    }
    __syncthreads();

    FragU A;
    const __bf16* ap = xb + rowA * CC + kk0;
    A.u[0] = *(const uint4*)(ap + hf * 8);
    A.u[1] = *(const uint4*)(ap + 16 + hf * 8);
    if (kk0 + 32 < CC) __builtin_prefetch(ap + 32, 0, 1);

    // Preload all 4 B fragments (8 ds_load_b128 in one clause, single wait),
    // then fire the 4 WMMAs back-to-back into the XDL pipe.
    const __bf16* bs = &smemB[p][0];
    FragU Bf[4];
#pragma unroll
    for (int nt = 0; nt < 4; ++nt) {
      const __bf16* bp = bs + (nt * 16 + ln) * 32 + hf * 8;
      Bf[nt].u[0] = *(const uint4*)(bp);
      Bf[nt].u[1] = *(const uint4*)(bp + 16);
    }
#pragma unroll
    for (int nt = 0; nt < 4; ++nt)
      acc[nt] = wmma_bf16(A.v, Bf[nt].v, acc[nt]);
    __syncthreads();
    p ^= 1;
  }

  const int sel = n0 / CC;                  // 0=Q 1=K 2=V
  const int hh  = (n0 % CC) / DD;
#pragma unroll
  for (int nt = 0; nt < 4; ++nt) {
    const int j  = n0 + nt * 16 + ln;
    const float bj = bias[j];
    const int d = nt * 16 + ln;             // n0 is 64-aligned -> d within head
#pragma unroll
    for (int v = 0; v < 8; ++v) {
      const int mg = mIdx * 16 + hf * 8 + v;
      const int b = mg / TT, t = mg % TT;
      const int bh = b * HH + hh;
      const __bf16 bv = (__bf16)(acc[nt][v] + bj);
      if (sel == 0)      qO[(bh * TT + t) * DD + d] = bv;
      else if (sel == 1) kO[(bh * TT + t) * DD + d] = bv;
      else               vtO[(bh * DD + d) * TT + t] = bv;
    }
  }
}

// ---------------------------------------------------------------- stage 3
// flash attention: one wave per 16-row Q tile, stream 32-wide K blocks
__global__ __launch_bounds__(256) void k_flash_attn(
    const __bf16* __restrict__ qI, const __bf16* __restrict__ kI,
    const __bf16* __restrict__ vtI, __bf16* __restrict__ yO) {
  __shared__ __align__(16) __bf16 smem[8][16 * 32];  // per-wave P staging
  const int lane  = threadIdx.x & 31;
  const int wslot = threadIdx.x >> 5;
  const int wave  = blockIdx.x * 8 + wslot;
  const int qTiles = TT / 16;               // 128
  const int bh = wave / qTiles;
  const int qb = (wave % qTiles) * 16;
  const int hf = lane >> 4, ln = lane & 15;

  const __bf16* qBase = qI + bh * TT * DD;
  const __bf16* kBase = kI + bh * TT * DD;
  const __bf16* vBase = vtI + bh * DD * TT;

  FragU Qf[2];                              // Q tile 16x64 = two K=32 A-frags
  {
    const __bf16* qp = qBase + (qb + ln) * DD;
    Qf[0].u[0] = *(const uint4*)(qp + hf * 8);
    Qf[0].u[1] = *(const uint4*)(qp + 16 + hf * 8);
    Qf[1].u[0] = *(const uint4*)(qp + 32 + hf * 8);
    Qf[1].u[1] = *(const uint4*)(qp + 48 + hf * 8);
  }

  f32x8 accO[4] = {};
  float rm[8], rl[8];
#pragma unroll
  for (int v = 0; v < 8; ++v) { rm[v] = -3.0e38f; rl[v] = 0.f; }

  const float sc = 0.18033688011112042f;    // (1/sqrt(64)) * log2(e)
  const int jBlocks = (qb + 16 + 31) / 32;

  for (int jb = 0; jb < jBlocks; ++jb) {
    const int k0 = jb * 32;
    if (jb + 1 < jBlocks) {
      __builtin_prefetch(kBase + (k0 + 32 + ln) * DD, 0, 1);
      __builtin_prefetch(vBase + ln * TT + k0 + 32, 0, 1);
    }
    // Preload all 4 K fragments, then 4 S-WMMAs back-to-back.
    FragU Kf[2][2];
#pragma unroll
    for (int st = 0; st < 2; ++st) {
      const __bf16* kp = kBase + (k0 + st * 16 + ln) * DD;
#pragma unroll
      for (int kc = 0; kc < 2; ++kc) {
        const __bf16* kpp = kp + kc * 32 + hf * 8;
        Kf[st][kc].u[0] = *(const uint4*)(kpp);
        Kf[st][kc].u[1] = *(const uint4*)(kpp + 16);
      }
    }
    f32x8 S[2] = {};
#pragma unroll
    for (int st = 0; st < 2; ++st)
#pragma unroll
      for (int kc = 0; kc < 2; ++kc)
        S[st] = wmma_bf16(Qf[kc].v, Kf[st][kc].v, S[st]);

    // Issue V-fragment global loads now; their latency hides under the
    // softmax VALU work (exp2 + cross-lane reductions) below.
    FragU Vf[4];
#pragma unroll
    for (int nt = 0; nt < 4; ++nt) {
      const __bf16* vp = vBase + (nt * 16 + ln) * TT + k0 + hf * 8;
      Vf[nt].u[0] = *(const uint4*)(vp);
      Vf[nt].u[1] = *(const uint4*)(vp + 16);
    }

    // scale + causal mask + per-row max over 32 cols
    float pmax[8];
#pragma unroll
    for (int v = 0; v < 8; ++v) {
      const int qi = qb + hf * 8 + v;
      float s0 = S[0][v] * sc, s1 = S[1][v] * sc;
      if (k0 + ln > qi)      s0 = -3.0e38f;
      if (k0 + 16 + ln > qi) s1 = -3.0e38f;
      S[0][v] = s0; S[1][v] = s1;
      pmax[v] = fmaxf(s0, s1);
    }
#pragma unroll
    for (int off = 1; off < 16; off <<= 1)
#pragma unroll
      for (int v = 0; v < 8; ++v)
        pmax[v] = fmaxf(pmax[v], __shfl_xor(pmax[v], off, 32));

    float alpha[8], psum[8];
#pragma unroll
    for (int v = 0; v < 8; ++v) {
      const float mNew = fmaxf(rm[v], pmax[v]);
      alpha[v] = exp2f(rm[v] - mNew);
      rm[v] = mNew;
      const float p0 = exp2f(S[0][v] - mNew);
      const float p1 = exp2f(S[1][v] - mNew);
      S[0][v] = p0; S[1][v] = p1;
      psum[v] = p0 + p1;
    }
#pragma unroll
    for (int off = 1; off < 16; off <<= 1)
#pragma unroll
      for (int v = 0; v < 8; ++v)
        psum[v] += __shfl_xor(psum[v], off, 32);
#pragma unroll
    for (int v = 0; v < 8; ++v) rl[v] = rl[v] * alpha[v] + psum[v];

    // C/D-layout P -> A-fragment layout via per-wave LDS tile
    __bf16* sw = smem[wslot];
#pragma unroll
    for (int v = 0; v < 8; ++v) {
      const int mrow = hf * 8 + v;
      sw[mrow * 32 + ln]      = (__bf16)S[0][v];
      sw[mrow * 32 + 16 + ln] = (__bf16)S[1][v];
#pragma unroll
      for (int nt = 0; nt < 4; ++nt) accO[nt][v] *= alpha[v];
    }
    __builtin_amdgcn_wave_barrier();
    FragU Pf;
    {
      const __bf16* sp = sw + ln * 32;
      Pf.u[0] = *(const uint4*)(sp + hf * 8);
      Pf.u[1] = *(const uint4*)(sp + 16 + hf * 8);
    }
    __builtin_amdgcn_wave_barrier();

#pragma unroll
    for (int nt = 0; nt < 4; ++nt)
      accO[nt] = wmma_bf16(Pf.v, Vf[nt].v, accO[nt]);
  }

  const int b = bh / HH, hh = bh % HH;
#pragma unroll
  for (int nt = 0; nt < 4; ++nt)
#pragma unroll
    for (int v = 0; v < 8; ++v) {
      const int t = qb + hf * 8 + v;
      yO[(b * TT + t) * CC + hh * DD + nt * 16 + ln] = (__bf16)(accO[nt][v] / rl[v]);
    }
}

// ---------------------------------------------------------------- stage 4
__global__ __launch_bounds__(256) void k_proj_gemm(
    const __bf16* __restrict__ yb, const __bf16* __restrict__ wpb,
    const float* __restrict__ bias, float* __restrict__ out) {
  __shared__ __align__(16) __bf16 smemB[2][64 * 32];
  const int lane  = threadIdx.x & 31;
  const int wslot = threadIdx.x >> 5;
  const int nGroups = CC / 64;              // 16
  const int mBlk = blockIdx.x / nGroups;
  const int n0   = (blockIdx.x % nGroups) * 64;
  const int mIdx = mBlk * 8 + wslot;
  const int hf = lane >> 4;
  const int ln = lane & 15;
  const int rowA = mIdx * 16 + ln;

  const __bf16* wTile = wpb + (size_t)n0 * CC;
  if (wslot == 0)
    tdm_load_tile32((unsigned)(uintptr_t)&smemB[0][0], wTile, 64u, (unsigned)CC);

  f32x8 acc[4] = {};
  int p = 0;
  constexpr int nIter = CC / 32;
  for (int it = 0; it < nIter; ++it) {
    const int kk0 = it * 32;
    if (wslot == 0) {
      if (it + 1 < nIter) {
        tdm_load_tile32((unsigned)(uintptr_t)&smemB[p ^ 1][0], wTile + kk0 + 32,
                        64u, (unsigned)CC);
        __builtin_amdgcn_s_wait_tensorcnt((short)1);
      } else {
        __builtin_amdgcn_s_wait_tensorcnt((short)0);
      }
    }
    __syncthreads();

    FragU A;
    const __bf16* ap = yb + rowA * CC + kk0;
    A.u[0] = *(const uint4*)(ap + hf * 8);
    A.u[1] = *(const uint4*)(ap + 16 + hf * 8);
    if (kk0 + 32 < CC) __builtin_prefetch(ap + 32, 0, 1);

    const __bf16* bs = &smemB[p][0];
    FragU Bf[4];
#pragma unroll
    for (int nt = 0; nt < 4; ++nt) {
      const __bf16* bp = bs + (nt * 16 + ln) * 32 + hf * 8;
      Bf[nt].u[0] = *(const uint4*)(bp);
      Bf[nt].u[1] = *(const uint4*)(bp + 16);
    }
#pragma unroll
    for (int nt = 0; nt < 4; ++nt)
      acc[nt] = wmma_bf16(A.v, Bf[nt].v, acc[nt]);
    __syncthreads();
    p ^= 1;
  }
#pragma unroll
  for (int nt = 0; nt < 4; ++nt) {
    const int j = n0 + nt * 16 + ln;
    const float bj = bias[j];
#pragma unroll
    for (int v = 0; v < 8; ++v) {
      const int mg = mIdx * 16 + hf * 8 + v;
      out[mg * CC + j] = acc[nt][v] + bj;
    }
  }
}

// ---------------------------------------------------------------- driver
extern "C" void kernel_launch(void* const* d_in, const int* in_sizes, int n_in,
                              void* d_out, int out_size, void* d_ws, size_t ws_size,
                              hipStream_t stream) {
  const float* x      = (const float*)d_in[0];
  const float* w_attn = (const float*)d_in[1];
  const float* b_attn = (const float*)d_in[2];
  const float* w_proj = (const float*)d_in[3];
  const float* b_proj = (const float*)d_in[4];
  float* out = (float*)d_out;

  char* ws = (char*)d_ws;
  size_t off = 0;
  auto carve = [&](size_t elems) {
    __bf16* p = (__bf16*)(ws + off);
    off += ((elems * 2 + 255) / 256) * 256;
    return p;
  };
  __bf16* xb  = carve((size_t)MM * CC);        // x in bf16
  __bf16* wab = carve((size_t)C3 * CC);        // w_attn bf16
  __bf16* wpb = carve((size_t)CC * CC);        // w_proj bf16
  __bf16* qB  = carve((size_t)MM * CC);        // Q [b,h,t,d]
  __bf16* kB  = carve((size_t)MM * CC);        // K [b,h,t,d]
  __bf16* vtB = carve((size_t)MM * CC);        // V [b,h,d,t]
  __bf16* ybB = carve((size_t)MM * CC);        // attn output bf16

  k_cvt_bf16<<<(MM * CC) / 1024, 256, 0, stream>>>(x, xb, MM * CC);
  k_cvt_bf16<<<(C3 * CC) / 1024, 256, 0, stream>>>(w_attn, wab, C3 * CC);
  k_cvt_bf16<<<(CC * CC) / 1024, 256, 0, stream>>>(w_proj, wpb, CC * CC);

  k_qkv_gemm<<<(MM / 128) * (C3 / 64), 256, 0, stream>>>(xb, wab, b_attn,
                                                         qB, kB, vtB);
  k_flash_attn<<<(BB * HH * (TT / 16)) / 8, 256, 0, stream>>>(qB, kB, vtB, ybB);
  k_proj_gemm<<<(MM / 128) * (CC / 64), 256, 0, stream>>>(ybB, wpb, b_proj, out);
}
